// HomoGNNModel_22582938042820
// MI455X (gfx1250) — matrix operand order repaired and checked by reading, not compiled, not run on hardware
//
#include <hip/hip_runtime.h>

typedef __bf16 bf16_t;
typedef __attribute__((ext_vector_type(16))) __bf16 v16bf;
typedef __attribute__((ext_vector_type(8)))  __bf16 v8bf;
typedef __attribute__((ext_vector_type(8)))  float   v8f;

#if __has_builtin(__builtin_amdgcn_sched_barrier)
#define SCHED_FENCE() __builtin_amdgcn_sched_barrier(0)
#else
#define SCHED_FENCE()
#endif

// ---------------------------------------------------------------------------
// Weight prep: f32 row-major [Wl; Wr] -> bf16 WMMA B-fragments, kc-major so a
// whole K-chunk (all N-tiles at one kc) is contiguous (async-DMA friendly):
//   frag[(((kc*Ntiles + nt)*32 + lane)*16 + j)] = W[k][n]
//   k = kc*32 + 8*(lane/16) + j + (j>=8 ? 8 : 0) ,  n = nt*16 + (lane%16)
// ---------------------------------------------------------------------------
__global__ void k_prep(const float* __restrict__ Wl, const float* __restrict__ Wr,
                       bf16_t* __restrict__ frag, int Kc, int Ntiles, int Nsrc) {
  int tid = blockIdx.x * blockDim.x + threadIdx.x;
  int total = Ntiles * Kc * 512;
  if (tid >= total) return;
  int j    = tid & 15;
  int lane = (tid >> 4) & 31;
  int rem  = tid >> 9;
  int nt   = rem % Ntiles;
  int kc   = rem / Ntiles;
  int half = lane >> 4, lr = lane & 15;
  int k = kc * 32 + 8 * half + j + ((j >= 8) ? 8 : 0);
  int n = nt * 16 + lr;
  float v = 0.0f;
  if (n < Nsrc) {
    if (k < 256) v = Wl[(size_t)k * Nsrc + n];
    else         v = Wr[(size_t)(k - 256) * Nsrc + n];
  }
  frag[tid] = (bf16_t)v;
}

// ---------------------------------------------------------------------------
// Aggregation: one wave per target; lane l owns columns [8l, 8l+8).
// FAN is compile-time so the gather loop fully unrolls into load clauses.
// Writes A[t] = [ bf16(mean_nbr) | bf16(x_tgt) ]  (512 cols).
// ---------------------------------------------------------------------------
template <int FAN>
__global__ __launch_bounds__(256)
void k_agg0(const float* __restrict__ x, const int* __restrict__ src,
            const int* __restrict__ tlid, bf16_t* __restrict__ A, int n_tgt) {
  int wave = (blockIdx.x * blockDim.x + threadIdx.x) >> 5;
  int lane = threadIdx.x & 31;
  if (wave >= n_tgt) return;
  const int c0 = lane * 8;
  float acc[8] = {};
  const long long eb = (long long)wave * FAN;
#pragma unroll
  for (int e = 0; e < FAN; ++e) {
    int s = src[eb + e];
    const float4* p = (const float4*)(x + (size_t)s * 256 + c0);
    float4 a = p[0], b = p[1];
    acc[0] += a.x; acc[1] += a.y; acc[2] += a.z; acc[3] += a.w;
    acc[4] += b.x; acc[5] += b.y; acc[6] += b.z; acc[7] += b.w;
  }
  const float inv = 1.0f / (float)FAN;
  bf16_t* d0 = A + (size_t)wave * 512 + c0;
#pragma unroll
  for (int i = 0; i < 8; ++i) d0[i] = (bf16_t)(acc[i] * inv);
  int tl = tlid[wave];
  const float4* q = (const float4*)(x + (size_t)tl * 256 + c0);
  float4 a = q[0], b = q[1];
  bf16_t* d1 = d0 + 256;
  d1[0] = (bf16_t)a.x; d1[1] = (bf16_t)a.y; d1[2] = (bf16_t)a.z; d1[3] = (bf16_t)a.w;
  d1[4] = (bf16_t)b.x; d1[5] = (bf16_t)b.y; d1[6] = (bf16_t)b.z; d1[7] = (bf16_t)b.w;
}

template <int FAN>
__global__ __launch_bounds__(256)
void k_agg1(const bf16_t* __restrict__ h, const int* __restrict__ src,
            const int* __restrict__ tlid, bf16_t* __restrict__ A, int n_tgt) {
  int wave = (blockIdx.x * blockDim.x + threadIdx.x) >> 5;
  int lane = threadIdx.x & 31;
  if (wave >= n_tgt) return;
  const int c0 = lane * 8;
  float acc[8] = {};
  const long long eb = (long long)wave * FAN;
#pragma unroll
  for (int e = 0; e < FAN; ++e) {
    int s = src[eb + e];
    v8bf r = *(const v8bf*)(h + (size_t)s * 256 + c0);
#pragma unroll
    for (int i = 0; i < 8; ++i) acc[i] += (float)r[i];
  }
  const float inv = 1.0f / (float)FAN;
  bf16_t* d0 = A + (size_t)wave * 512 + c0;
#pragma unroll
  for (int i = 0; i < 8; ++i) d0[i] = (bf16_t)(acc[i] * inv);
  int tl = tlid[wave];
  v8bf r = *(const v8bf*)(h + (size_t)tl * 256 + c0);
  bf16_t* d1 = d0 + 256;
#pragma unroll
  for (int i = 0; i < 8; ++i) d1[i] = r[i];
}

// ---------------------------------------------------------------------------
// Async global->LDS staging (CDNA5): 16B per lane per issue, tracked by
// ASYNCcnt. LDS offset = low 32 bits of the flat shared address.
// ---------------------------------------------------------------------------
__device__ __forceinline__ void async_b128_to_lds(void* lds_ptr, const void* gptr) {
  uint32_t lds_off = (uint32_t)(uintptr_t)lds_ptr;
  asm volatile("global_load_async_to_lds_b128 %0, %1, off"
               :: "v"(lds_off), "v"(gptr) : "memory");
}
__device__ __forceinline__ void wait_async0() {
  asm volatile("s_wait_asynccnt 0x0" ::: "memory");
}

// ---------------------------------------------------------------------------
// Fused SAGE GEMM + bias + LayerNorm + ReLU.
// Block = 8 waves; wave owns a 16-row x 256-col output stripe.
// B K-chunks (16 KB) double-buffered in LDS via async DMA. Inside a K-step,
// B fragments run a distance-1 register pipeline; sched_barrier(0) after each
// WMMA bounds scheduler lookahead so only 2 fragments are ever live (keeps
// the wave under 256 VGPRs -> no bank-switch/copy churn) while still hiding
// ds_load latency under the previous WMMA.
// ---------------------------------------------------------------------------
#define CHUNK_BYTES 16384
__global__ __launch_bounds__(256)
void k_sage(const bf16_t* __restrict__ A, const bf16_t* __restrict__ Bfrag,
            const float* __restrict__ bias, const float* __restrict__ g,
            const float* __restrict__ be, bf16_t* __restrict__ out, int M) {
  __shared__ __align__(32) bf16_t sB[2][CHUNK_BYTES / 2];
  const int tid  = threadIdx.x;
  const int w    = tid >> 5;
  const int lane = tid & 31;
  const int half = lane >> 4, lr = lane & 15;
  const int m0   = (blockIdx.x * 8 + w) * 16;   // M is a multiple of 128

  auto stage = [&](int kc, int b) {
    const char* gsrc = (const char*)Bfrag + (size_t)kc * CHUNK_BYTES;
    char* lbase = (char*)&sB[b][0];
#pragma unroll
    for (int i = 0; i < 4; ++i) {
      uint32_t off = (uint32_t)(i * 4096 + tid * 16);
      async_b128_to_lds(lbase + off, gsrc + off);
    }
  };

  const bf16_t* arow = A + (size_t)(m0 + lr) * 512;
  auto loadA = [&](int kc) -> v16bf {
    const int kb = kc * 32 + half * 8;
    v8bf r0 = *(const v8bf*)(arow + kb);
    v8bf r1 = *(const v8bf*)(arow + kb + 16);
    v16bf a;
#pragma unroll
    for (int i = 0; i < 8; ++i) { a[i] = r0[i]; a[i + 8] = r1[i]; }
    return a;
  };

  stage(0, 0);
  v16bf a = loadA(0);
  v8f acc[16] = {};

  for (int kc = 0; kc < 16; ++kc) {
    wait_async0();          // own async portion of chunk kc complete
    __syncthreads();        // all portions visible; prev compute buffer free
    if (kc + 1 < 16) stage(kc + 1, (kc + 1) & 1);
    v16bf anext = (kc + 1 < 16) ? loadA(kc + 1) : a;

    const bf16_t* bbase = &sB[kc & 1][lane * 16];
    v16bf bcur = *(const v16bf*)(bbase);
#pragma unroll
    for (int nt = 0; nt < 16; ++nt) {
      v16bf bnext = (nt < 15) ? *(const v16bf*)(bbase + (size_t)(nt + 1) * 512)
                              : bcur;
      acc[nt] = __builtin_amdgcn_wmma_f32_16x16x32_bf16(
          false, a, false, bcur, (short)0, acc[nt], false, false);
      SCHED_FENCE();        // cap lookahead: max one B fragment in flight
      bcur = bnext;
    }
    a = anext;
  }

  float bv[16], gv[16], bev[16];
#pragma unroll
  for (int nt = 0; nt < 16; ++nt) {
    int c = nt * 16 + lr;
    bv[nt] = bias[c]; gv[nt] = g[c]; bev[nt] = be[c];
  }
#pragma unroll
  for (int v = 0; v < 8; ++v) {
    float s = 0.0f, s2 = 0.0f;
#pragma unroll
    for (int nt = 0; nt < 16; ++nt) {
      float t = acc[nt][v] + bv[nt];
      s += t; s2 += t * t;
    }
    for (int msk = 1; msk < 16; msk <<= 1) {
      s  += __shfl_xor(s,  msk, 32);
      s2 += __shfl_xor(s2, msk, 32);
    }
    float mu  = s * (1.0f / 256.0f);
    float var = s2 * (1.0f / 256.0f) - mu * mu;
    float rs  = rsqrtf(var + 1e-5f);
    bf16_t* orow = out + (size_t)(m0 + v + 8 * half) * 256;
#pragma unroll
    for (int nt = 0; nt < 16; ++nt) {
      float t = acc[nt][v] + bv[nt];
      float o = (t - mu) * rs * gv[nt] + bev[nt];
      o = fmaxf(o, 0.0f);
      orow[nt * 16 + lr] = (bf16_t)o;
    }
  }
}

// ---------------------------------------------------------------------------
// FC head: [4096,256] @ [256,153] + bias -> f32. N padded to 160 (10 tiles).
// Tiny (32 blocks); weights stay L2-resident, one-ahead fragment pipelining.
// ---------------------------------------------------------------------------
__global__ __launch_bounds__(256)
void k_fc(const bf16_t* __restrict__ A, const bf16_t* __restrict__ Bfrag,
          const float* __restrict__ fcb, float* __restrict__ out, int M) {
  const int w    = threadIdx.x >> 5;
  const int lane = threadIdx.x & 31;
  const int half = lane >> 4, lr = lane & 15;
  const int m0   = (blockIdx.x * 8 + w) * 16;
  if (m0 >= M) return;

  v8f acc[10] = {};
  const bf16_t* arow = A + (size_t)(m0 + lr) * 256;

  for (int kc = 0; kc < 8; ++kc) {
    const int kb = kc * 32 + half * 8;
    v8bf r0 = *(const v8bf*)(arow + kb);
    v8bf r1 = *(const v8bf*)(arow + kb + 16);
    v16bf a;
#pragma unroll
    for (int i = 0; i < 8; ++i) { a[i] = r0[i]; a[i + 8] = r1[i]; }

    const bf16_t* bbase = Bfrag + ((size_t)kc * 10 * 32 + lane) * 16;
    v16bf bcur = *(const v16bf*)(bbase);
#pragma unroll
    for (int nt = 0; nt < 10; ++nt) {
      v16bf bnext = (nt < 9) ? *(const v16bf*)(bbase + (size_t)(nt + 1) * 512)
                             : bcur;
      acc[nt] = __builtin_amdgcn_wmma_f32_16x16x32_bf16(
          false, a, false, bcur, (short)0, acc[nt], false, false);
      SCHED_FENCE();
      bcur = bnext;
    }
  }

#pragma unroll
  for (int v = 0; v < 8; ++v) {
    float* orow = out + (size_t)(m0 + v + 8 * half) * 153;
#pragma unroll
    for (int nt = 0; nt < 10; ++nt) {
      int col = nt * 16 + lr;
      if (col < 153) orow[col] = acc[nt][v] + fcb[col];
    }
  }
}

// ---------------------------------------------------------------------------
extern "C" void kernel_launch(void* const* d_in, const int* in_sizes, int n_in,
                              void* d_out, int out_size, void* d_ws, size_t ws_size,
                              hipStream_t stream) {
  const float* x_feat = (const float*)d_in[0];
  const int*   e0_src = (const int*)d_in[1];
  const int*   tlid0  = (const int*)d_in[3];
  const int*   e1_src = (const int*)d_in[4];
  const int*   tlid1  = (const int*)d_in[6];
  const float* Wl0 = (const float*)d_in[7];
  const float* Wr0 = (const float*)d_in[8];
  const float* b0  = (const float*)d_in[9];
  const float* g0  = (const float*)d_in[10];
  const float* be0 = (const float*)d_in[11];
  const float* Wl1 = (const float*)d_in[12];
  const float* Wr1 = (const float*)d_in[13];
  const float* b1  = (const float*)d_in[14];
  const float* g1  = (const float*)d_in[15];
  const float* be1 = (const float*)d_in[16];
  const float* fcW = (const float*)d_in[17];
  const float* fcb = (const float*)d_in[18];
  float* out = (float*)d_out;

  const int N1 = 45056, N2 = 4096;

  char* ws = (char*)d_ws;
  size_t off = 0;
  auto carve = [&](size_t bytes) -> char* {
    char* p = ws + off;
    off = (off + bytes + 255) & ~(size_t)255;
    return p;
  };
  bf16_t* W0f = (bf16_t*)carve((size_t)512 * 256 * 2);
  bf16_t* W1f = (bf16_t*)carve((size_t)512 * 256 * 2);
  bf16_t* Wff = (bf16_t*)carve((size_t)256 * 160 * 2);
  bf16_t* A0  = (bf16_t*)carve((size_t)N1 * 512 * 2);
  bf16_t* H1  = (bf16_t*)carve((size_t)N1 * 256 * 2);
  bf16_t* A1  = (bf16_t*)carve((size_t)N2 * 512 * 2);
  bf16_t* H2  = (bf16_t*)carve((size_t)N2 * 256 * 2);

  // Weight fragment prep (tiny)
  k_prep<<<(512 * 256 + 255) / 256, 256, 0, stream>>>(Wl0, Wr0, W0f, 16, 16, 256);
  k_prep<<<(512 * 256 + 255) / 256, 256, 0, stream>>>(Wl1, Wr1, W1f, 16, 16, 256);
  k_prep<<<(256 * 160 + 255) / 256, 256, 0, stream>>>(fcW, fcW, Wff, 8, 10, 153);

  // Layer 0
  k_agg0<15><<<N1 / 8, 256, 0, stream>>>(x_feat, e0_src, tlid0, A0, N1);
  k_sage<<<N1 / 128, 256, 0, stream>>>(A0, W0f, b0, g0, be0, H1, N1);

  // Layer 1
  k_agg1<10><<<N2 / 8, 256, 0, stream>>>(H1, e1_src, tlid1, A1, N2);
  k_sage<<<N2 / 128, 256, 0, stream>>>(A1, W1f, b1, g1, be1, H2, N2);

  // Head
  k_fc<<<N2 / 128, 256, 0, stream>>>(H2, Wff, fcb, out, N2);
}
